// AttentionModule_56959856280123
// MI455X (gfx1250) — matrix operand order repaired and checked by reading, not compiled, not run on hardware
//
#include <hip/hip_runtime.h>
#include <hip/hip_bf16.h>
#include <math.h>

typedef __attribute__((ext_vector_type(16))) __bf16 v16bf;
typedef __attribute__((ext_vector_type(8)))  __bf16 v8bf;
typedef __attribute__((ext_vector_type(8)))  float  v8f;

#define BB 8
#define CC 128
#define HH 256
#define WW 256
#define HWSZ 65536

// LDS row strides (in halves / floats) chosen for 16B alignment + bank spread
#define WLD 136   // weight / xT rows (128 data + 8 pad)
#define KLD 264   // K / KT / Q / P rows (256 data + 8 pad)
#define SLD 260   // score rows (f32)

// ---- WMMA fragment loaders (ISA 7.12.2 layouts, wave32) -------------------
// A (16x32 bf16): lane<16 -> row=lane, K={0..7,16..23}; lane>=16 -> row=lane-16,
// K={8..15,24..31}. Two contiguous 16B runs per lane.
__device__ __forceinline__ v16bf frag_a(const __bf16* row0, int ld, int lane) {
  const int r  = lane & 15;
  const int ko = (lane & 16) ? 8 : 0;
  const __bf16* p = row0 + r * ld + ko;
  v8bf lo = *(const v8bf*)(p);
  v8bf hi = *(const v8bf*)(p + 16);
  v16bf f;
#pragma unroll
  for (int i = 0; i < 8; ++i) { f[i] = lo[i]; f[i + 8] = hi[i]; }
  return f;
}

// B (32x16 bf16): lane<16 -> col N=lane holds K={0..7,16..23}; lane>=16 -> N=lane-16
// holds K={8..15,24..31}. Two 16B runs from the N-major (transposed) copy.
__device__ __forceinline__ v16bf frag_b(const __bf16* rowT0, int ld, int lane) {
  const int n  = lane & 15;
  const int ko = (lane & 16) ? 8 : 0;
  const __bf16* p = rowT0 + n * ld + ko;
  v8bf lo = *(const v8bf*)(p);
  v8bf hi = *(const v8bf*)(p + 16);
  v16bf f;
#pragma unroll
  for (int i = 0; i < 8; ++i) { f[i] = lo[i]; f[i + 8] = hi[i]; }
  return f;
}

// Async global->LDS 16B copy (CDNA5, ASYNCcnt-tracked, no VGPR data path).
__device__ __forceinline__ void async_copy_b128(unsigned lds_off, const void* gaddr) {
  asm volatile("global_load_async_to_lds_b128 %0, %1, off"
               :: "v"(lds_off), "v"(gaddr) : "memory");
}
__device__ __forceinline__ void wait_async0() {
  asm volatile("s_wait_asynccnt 0x0" ::: "memory");
}

// ---- Kernel 1: q = W1@x + b1, k = W2@x + b2, stored bf16 ------------------
// grid.x = B * (HW/128); block = 256 threads (8 waves)
__global__ __launch_bounds__(256) void qk_kernel(
    const float* __restrict__ x,
    const float* __restrict__ W1, const float* __restrict__ b1,
    const float* __restrict__ W2, const float* __restrict__ b2,
    __bf16* __restrict__ q, __bf16* __restrict__ k)
{
  extern __shared__ char smem[];
  __bf16* w1s = (__bf16*)smem;             // 128 x WLD   (row-major [o][c])
  __bf16* w2s = w1s + 128 * WLD;
  __bf16* xT  = w2s + 128 * WLD;           // 128 x WLD   (pos-major [s][c])
  float*  bsh = (float*)(xT + 128 * WLD);  // 256 floats  (b1 | b2)

  const int tid  = threadIdx.x;
  const int lane = tid & 31;
  const int wave = tid >> 5;
  const int bidx = blockIdx.x >> 9;                 // batch
  const long s0  = (long)(blockIdx.x & 511) * 128;  // spatial tile base

  for (int idx = tid; idx < 128 * 128; idx += 256) {
    int o = idx >> 7, c = idx & 127;
    w1s[o * WLD + c] = (__bf16)W1[idx];
    w2s[o * WLD + c] = (__bf16)W2[idx];
  }
  if (tid < 128) { bsh[tid] = b1[tid]; bsh[128 + tid] = b2[tid]; }
  const float* xb = x + (long)bidx * CC * HWSZ;
  for (int idx = tid; idx < 128 * 128; idx += 256) {
    int c = idx >> 7, s = idx & 127;                 // coalesced in s
    xT[s * WLD + c] = (__bf16)xb[(long)c * HWSZ + s0 + s];
  }
  __syncthreads();

  const __bf16* bT = xT + wave * 16 * WLD;           // this wave's 16 positions
  const int hif = (lane >> 4) & 1;
  const int n   = lane & 15;
  const long scol = s0 + wave * 16 + n;

  // Hoist the 4 B fragments (x tile) -- shared by all 8 output tiles & both GEMMs
  v16bf bF[4];
#pragma unroll
  for (int kc = 0; kc < 4; ++kc) bF[kc] = frag_b(bT + kc * 32, WLD, lane);

#pragma unroll
  for (int mt = 0; mt < 8; ++mt) {                   // 8 output-channel tiles
    v8f aq = {}; v8f ak = {};
#pragma unroll
    for (int kc = 0; kc < 4; ++kc) {                 // C=128 -> 4 k-steps of 32
      v16bf a1 = frag_a(w1s + mt * 16 * WLD + kc * 32, WLD, lane);
      v16bf a2 = frag_a(w2s + mt * 16 * WLD + kc * 32, WLD, lane);
      aq = __builtin_amdgcn_wmma_f32_16x16x32_bf16(false, a1, false, bF[kc], (short)0, aq, false, false);
      ak = __builtin_amdgcn_wmma_f32_16x16x32_bf16(false, a2, false, bF[kc], (short)0, ak, false, false);
    }
#pragma unroll
    for (int v = 0; v < 8; ++v) {                    // C layout: M=v+8*hif, N=n
      int o = mt * 16 + v + 8 * hif;
      long base = ((long)(bidx * CC + o)) * HWSZ + scol;
      q[base] = (__bf16)(aq[v] + bsh[o]);
      k[base] = (__bf16)(ak[v] + bsh[128 + o]);
    }
  }
}

// ---- Kernel 2: fused S=Q*K^T -> softmax -> O=P*K -> +x --------------------
// grid = (B*C, H/32); block = 256 threads (8 waves). LDS ~314KB (CDNA5 WGP).
__global__ __launch_bounds__(256) void attn_kernel(
    const float* __restrict__ x,
    const __bf16* __restrict__ q, const __bf16* __restrict__ k,
    float* __restrict__ out)
{
  extern __shared__ char smem[];
  __bf16* Ks   = (__bf16*)smem;              // 256 x KLD  (row-major [g][w])
  __bf16* KTs  = Ks + 256 * KLD;             // 256 x KLD  ([w][g])
  __bf16* QP   = KTs + 256 * KLD;            // 32 x KLD   (Q, then aliased P)
  float*  Ss   = (float*)(QP + 32 * KLD);    // 32 x SLD   (scores, f32)
  float*  red  = Ss + 32 * SLD;              // 256        (row reductions)
  float*  rinv = red + 256;                  // 32         (1/rowsum)

  const int tid  = threadIdx.x;
  const int lane = tid & 31;
  const int wave = tid >> 5;
  const int bc   = blockIdx.x;               // b*C + c
  const int h0   = blockIdx.y * 32;

  const __bf16* Kg = k + (long)bc * HWSZ;
  const __bf16* Qg = q + (long)bc * HWSZ;
  const float*  xb = x + (long)bc * HWSZ;
  float*        ob = out + (long)bc * HWSZ;

  // --- Async-stage K (256x256) and Q (32x256) straight into LDS (ASYNCcnt) ---
  {
    const unsigned ksBase = (unsigned)(unsigned long long)(void*)Ks;
    for (int g = wave; g < 256; g += 8) {
      unsigned lo = ksBase + (unsigned)(g * KLD + lane * 8) * 2u;
      async_copy_b128(lo, (const void*)(Kg + (long)g * 256 + lane * 8));
    }
    const unsigned qBase = (unsigned)(unsigned long long)(void*)QP;
    for (int r = wave; r < 32; r += 8) {
      unsigned lo = qBase + (unsigned)(r * KLD + lane * 8) * 2u;
      async_copy_b128(lo, (const void*)(Qg + (long)(h0 + r) * 256 + lane * 8));
    }
  }
  // Warm the residual x tile (32 rows x 256 f32) while the DMA runs.
  __builtin_prefetch(xb + (long)(h0 + (tid >> 3)) * 256 + (tid & 7) * 32, 0, 0);
  wait_async0();
  __syncthreads();

  // Build KT (w-major) from the LDS copy of K
  for (int g = wave; g < 256; g += 8) {
    v8bf kv = *(const v8bf*)(Ks + g * KLD + lane * 8);
#pragma unroll
    for (int j = 0; j < 8; ++j) KTs[(lane * 8 + j) * KLD + g] = kv[j];
  }
  __syncthreads();

  const int rt   = wave >> 2;     // row tile (0..1)
  const int csub = wave & 3;      // col-tile phase
  const int r0   = rt * 16;
  const int hif  = (lane >> 4) & 1;
  const int n    = lane & 15;

  // S = Q * K^T : hoist the 8 Q-row A fragments, sweep 4 column tiles
  {
    v16bf aF[8];
#pragma unroll
    for (int ks = 0; ks < 8; ++ks) aF[ks] = frag_a(QP + r0 * KLD + ks * 32, KLD, lane);
#pragma unroll
    for (int ci = 0; ci < 4; ++ci) {
      const int ct = csub + ci * 4;
      v8f acc = {};
#pragma unroll
      for (int ks = 0; ks < 8; ++ks) {
        v16bf b = frag_b(Ks + ct * 16 * KLD + ks * 32, KLD, lane);
        acc = __builtin_amdgcn_wmma_f32_16x16x32_bf16(false, aF[ks], false, b, (short)0, acc, false, false);
      }
#pragma unroll
      for (int v = 0; v < 8; ++v)
        Ss[(r0 + v + 8 * hif) * SLD + ct * 16 + n] = acc[v];
    }
  }
  __syncthreads();

  // Row softmax (f32), unnormalized exp -> P (bf16, aliases dead Q buffer)
  {
    const int r  = tid & 31;
    const int ch = tid >> 5;                 // 8 chunks of 32 cols
    const float* srow = Ss + r * SLD + ch * 32;
    float m = -3.0e38f;
#pragma unroll
    for (int j = 0; j < 32; ++j) m = fmaxf(m, srow[j]);
    red[r * 8 + ch] = m;
    __syncthreads();
    float rm = -3.0e38f;
#pragma unroll
    for (int j = 0; j < 8; ++j) rm = fmaxf(rm, red[r * 8 + j]);
    __syncthreads();                         // all max reads done before reuse
    float s = 0.f;
    __bf16* prow = QP + r * KLD + ch * 32;
#pragma unroll
    for (int j = 0; j < 32; ++j) {
      float e = __expf(srow[j] - rm);
      s += e;
      prow[j] = (__bf16)e;
    }
    red[r * 8 + ch] = s;
    __syncthreads();
    if (ch == 0) {
      float tot = 0.f;
#pragma unroll
      for (int j = 0; j < 8; ++j) tot += red[r * 8 + j];
      rinv[r] = 1.0f / tot;
    }
  }
  __syncthreads();

  // O = P * K : hoist the 8 P-row A fragments; B from KT.
  // Fold the 1/rowsum scale and the residual add into the epilogue.
  {
    v16bf aF[8];
#pragma unroll
    for (int ks = 0; ks < 8; ++ks) aF[ks] = frag_a(QP + r0 * KLD + ks * 32, KLD, lane);
#pragma unroll
    for (int ci = 0; ci < 4; ++ci) {
      const int ct = csub + ci * 4;
      v8f acc = {};
#pragma unroll
      for (int ks = 0; ks < 8; ++ks) {
        v16bf b = frag_b(KTs + ct * 16 * KLD + ks * 32, KLD, lane);
        acc = __builtin_amdgcn_wmma_f32_16x16x32_bf16(false, aF[ks], false, b, (short)0, acc, false, false);
      }
#pragma unroll
      for (int v = 0; v < 8; ++v) {
        int r = r0 + v + 8 * hif;
        long idx = (long)(h0 + r) * 256 + ct * 16 + n;
        ob[idx] = acc[v] * rinv[r] + xb[idx];
      }
    }
  }
}

extern "C" void kernel_launch(void* const* d_in, const int* in_sizes, int n_in,
                              void* d_out, int out_size, void* d_ws, size_t ws_size,
                              hipStream_t stream) {
  (void)in_sizes; (void)n_in; (void)out_size; (void)ws_size;
  const float* x  = (const float*)d_in[0];
  const float* W1 = (const float*)d_in[1];
  const float* b1 = (const float*)d_in[2];
  const float* W2 = (const float*)d_in[3];
  const float* b2 = (const float*)d_in[4];
  float* out = (float*)d_out;

  __bf16* qws = (__bf16*)d_ws;                       // 128 MB
  __bf16* kws = qws + (size_t)BB * CC * HWSZ;        // 128 MB

  const size_t lds1 = (size_t)3 * 128 * WLD * sizeof(__bf16) + 256 * sizeof(float);
  qk_kernel<<<dim3(BB * 512), dim3(256), lds1, stream>>>(x, W1, b1, W2, b2, qws, kws);

  const size_t lds2 = (size_t)2 * 256 * KLD * sizeof(__bf16)  // K + KT
                    + (size_t)32 * KLD * sizeof(__bf16)       // Q/P
                    + (size_t)32 * SLD * sizeof(float)        // S
                    + 256 * sizeof(float) + 32 * sizeof(float);
  attn_kernel<<<dim3(BB * CC, HH / 32), dim3(256), lds2, stream>>>(x, qws, kws, out);
}
